// Block0_64029372448996
// MI455X (gfx1250) — matrix operand order repaired and checked by reading, not compiled, not run on hardware
//
#include <hip/hip_runtime.h>
#include <hip/hip_bf16.h>

typedef _Float16 f16;
typedef __attribute__((ext_vector_type(16))) _Float16 v16h;
typedef __attribute__((ext_vector_type(8)))  _Float16 v8h;
typedef __attribute__((ext_vector_type(8)))  float    v8f;

#define CC   96
#define HH   64
#define WW   128
#define HWS  (HH*WW)          // 8192
#define NPOS (4*HWS)          // 32768 output positions

__device__ __forceinline__ v16h cat16(v8h lo, v8h hi) {
    return __builtin_shufflevector(lo, hi, 0,1,2,3,4,5,6,7,8,9,10,11,12,13,14,15);
}
__device__ __forceinline__ v8f wmma16(v16h a, v16h b, v8f c) {
    // D = A(16x32 f16) * B(32x16 f16) + C(16x16 f32)
    return __builtin_amdgcn_wmma_f32_16x16x32_f16(false, a, false, b, (short)0, c, false, false);
}

// Branch-free erf (Abramowitz-Stegun 7.1.26, |err| < 1.5e-7): one v_exp_f32,
// one v_rcp_f32, 6 FMAs, no EXEC-divergent paths inside the WMMA loop.
__device__ __forceinline__ float erf_fast(float x) {
    float ax = fabsf(x);
    float t  = __builtin_amdgcn_rcpf(fmaf(0.3275911f, ax, 1.0f));
    float p  = fmaf(t, 1.061405429f, -1.453152027f);
    p = fmaf(t, p, 1.421413741f);
    p = fmaf(t, p, -0.284496736f);
    p = fmaf(t, p, 0.254829592f);
    p = p * t;
    float e = __expf(-ax * ax);
    float r = fmaf(-p, e, 1.0f);
    return copysignf(r, x);
}
__device__ __forceinline__ float gelu_f(float v) {
    return 0.5f * v * (1.0f + erf_fast(v * 0.70710678118654752440f));
}

// ---------------------------------------------------------------------------
// Kernel: per-batch tiled transpose  src[R][C] -> dst[C][R]
// ---------------------------------------------------------------------------
__global__ __launch_bounds__(256) void k_transpose(const float* __restrict__ src,
                                                   float* __restrict__ dst,
                                                   int R, int C) {
    __shared__ float tile[32][33];
    const int b  = blockIdx.z;
    const size_t base = (size_t)b * R * C;
    const int c0 = blockIdx.x * 32, r0 = blockIdx.y * 32;
    const int tx = threadIdx.x & 31, ty = threadIdx.x >> 5;
#pragma unroll
    for (int i = 0; i < 32; i += 8) {
        int r = r0 + ty + i, c = c0 + tx;
        if (r < R && c < C) tile[ty + i][tx] = src[base + (size_t)r * C + c];
    }
    __syncthreads();
#pragma unroll
    for (int i = 0; i < 32; i += 8) {
        int c = c0 + ty + i, r = r0 + tx;
        if (r < R && c < C) dst[base + (size_t)c * R + r] = tile[tx][ty + i];
    }
}

// ---------------------------------------------------------------------------
// Kernel: pack pw1_w (384x96) and pw2_w (96x384) into WMMA B-fragment layout.
// B (32x16 f16): lanes 0-15 -> N=lane, K=kb*32+{0..7,16..23} ; lanes 16-31 -> K=kb*32+{8..15,24..31}
// Storage: [ntile][kblock][lane][16 halves]
// ---------------------------------------------------------------------------
__global__ __launch_bounds__(256) void k_pack(const float* __restrict__ pw1_w,
                                              const float* __restrict__ pw2_w,
                                              f16* __restrict__ b1f,
                                              f16* __restrict__ b2f) {
    int tid = blockIdx.x * 256 + threadIdx.x;
    if (tid < 24 * 3 * 32) {            // GEMM1 B: N=384 (24 tiles), K=96 (3 blocks)
        int lane = tid & 31, kb = (tid >> 5) % 3, nt = tid / 96;
        int n = nt * 16 + (lane & 15);
        int kbase = kb * 32 + (lane >> 4) * 8;
        f16* o = b1f + (size_t)tid * 16;
#pragma unroll
        for (int e = 0; e < 16; ++e) o[e] = (f16)pw1_w[(size_t)n * 96 + kbase + e + (e >= 8 ? 8 : 0)];
    } else if (tid < 24 * 3 * 32 + 6 * 12 * 32) {  // GEMM2 B: N=96 (6 tiles), K=384 (12 blocks)
        int t = tid - 24 * 3 * 32;
        int lane = t & 31, kb = (t >> 5) % 12, nt = t / 384;
        int n = nt * 16 + (lane & 15);
        int kbase = kb * 32 + (lane >> 4) * 8;
        f16* o = b2f + (size_t)t * 16;
#pragma unroll
        for (int e = 0; e < 16; ++e) o[e] = (f16)pw2_w[(size_t)n * 384 + kbase + e + (e >= 8 ? 8 : 0)];
    }
}

// ---------------------------------------------------------------------------
// Kernel: fused bilinear-LUT sampling + stride-7 depthwise 7x7 conv + bias
//         + LayerNorm, emitting f16 activations row-major [NPOS][96].
// 1 wave = 1 output position; lane handles channels {l, l+32, l+64}.
// ---------------------------------------------------------------------------
__global__ __launch_bounds__(128) void k_bilin_dw_ln(const float* __restrict__ xt,   // BHWC
                                                     const float* __restrict__ lut1,
                                                     const float* __restrict__ lut2,
                                                     const float* __restrict__ dw_w,
                                                     const float* __restrict__ dw_b,
                                                     const float* __restrict__ norm_w,
                                                     const float* __restrict__ norm_b,
                                                     f16* __restrict__ ynorm) {
    __shared__ float sdw[CC * 49];
    const int tid = threadIdx.x;
    for (int i = tid; i < CC * 49; i += 128) sdw[i] = dw_w[i];
    __syncthreads();

    const int lane = tid & 31;
    const int wv   = tid >> 5;
    const int p    = blockIdx.x * 4 + wv;       // 0..32767
    const int b    = p >> 13;
    const int s    = p & 8191;
    const int oh   = s >> 7;
    const int ow   = s & 127;
    const float* lut = (ow < 64) ? lut1 : lut2; // width-concat split at 64*7

    // preload this position's 49 tap coordinates across lanes
    float cx0, cy0, cx1 = 0.f, cy1 = 0.f;
    {
        int t = lane, kh = t / 7, kw = t - kh * 7;
        int n = (oh * 7 + kh) * (WW * 7) + (ow * 7 + kw);
        cx0 = lut[2 * n]; cy0 = lut[2 * n + 1];
    }
    if (lane + 32 < 49) {
        int t = lane + 32, kh = t / 7, kw = t - kh * 7;
        int n = (oh * 7 + kh) * (WW * 7) + (ow * 7 + kw);
        cx1 = lut[2 * n]; cy1 = lut[2 * n + 1];
    }

    const int c0 = lane, c1 = lane + 32, c2 = lane + 64;
    float a0 = 0.f, a1 = 0.f, a2 = 0.f;

    for (int t = 0; t < 49; ++t) {
        int src = t & 31;
        float cx = (t < 32) ? __shfl(cx0, src, 32) : __shfl(cx1, src, 32);
        float cy = (t < 32) ? __shfl(cy0, src, 32) : __shfl(cy1, src, 32);
        int x1i = min(max((int)floorf(cx), 0), WW - 1);
        int x2i = min(max((int)floorf(cx) + 1, 0), WW - 1);
        int y1i = min(max((int)floorf(cy), 0), HH - 1);
        int y2i = min(max((int)floorf(cy) + 1, 0), HH - 1);
        float dx1 = cx - (float)x1i, dx2 = (float)x2i - cx;
        float dy1 = cy - (float)y1i, dy2 = (float)y2i - cy;
        float w11 = dx2 * dy2, w12 = dx1 * dy2, w21 = dx2 * dy1, w22 = dx1 * dy1;

        const size_t pb = (size_t)b * HWS;
        const float* p11 = xt + (pb + (size_t)y1i * WW + x1i) * CC;
        const float* p12 = xt + (pb + (size_t)y1i * WW + x2i) * CC;
        const float* p21 = xt + (pb + (size_t)y2i * WW + x1i) * CC;
        const float* p22 = xt + (pb + (size_t)y2i * WW + x2i) * CC;

        float v0 = w11 * p11[c0] + w12 * p12[c0] + w21 * p21[c0] + w22 * p22[c0];
        float v1 = w11 * p11[c1] + w12 * p12[c1] + w21 * p21[c1] + w22 * p22[c1];
        float v2 = w11 * p11[c2] + w12 * p12[c2] + w21 * p21[c2] + w22 * p22[c2];
        a0 += sdw[c0 * 49 + t] * v0;
        a1 += sdw[c1 * 49 + t] * v1;
        a2 += sdw[c2 * 49 + t] * v2;
    }
    a0 += dw_b[c0]; a1 += dw_b[c1]; a2 += dw_b[c2];

    // wave32 LayerNorm over 96 channels
    float sum = a0 + a1 + a2;
    float sq  = a0 * a0 + a1 * a1 + a2 * a2;
#pragma unroll
    for (int m = 16; m >= 1; m >>= 1) {
        sum += __shfl_xor(sum, m, 32);
        sq  += __shfl_xor(sq,  m, 32);
    }
    const float mean = sum * (1.0f / 96.0f);
    const float var  = sq * (1.0f / 96.0f) - mean * mean;
    const float rstd = rsqrtf(var + 1e-6f);

    f16* yo = ynorm + (size_t)p * CC;
    yo[c0] = (f16)((a0 - mean) * rstd * norm_w[c0] + norm_b[c0]);
    yo[c1] = (f16)((a1 - mean) * rstd * norm_w[c1] + norm_b[c1]);
    yo[c2] = (f16)((a2 - mean) * rstd * norm_w[c2] + norm_b[c2]);
}

// ---------------------------------------------------------------------------
// Kernel: fused MLP  (96 -> 384 -> GELU -> 384 -> 96) + gamma scale + residual.
// 8 waves/block; each wave owns 16 rows. GEMM1: 24 n-tiles x 3 WMMAs,
// GEMM2: 6 n-tiles x 12 WMMAs, all v_wmma_f32_16x16x32_f16.
// Residual applied in-place on the BHWC buffer.
// ---------------------------------------------------------------------------
__global__ __launch_bounds__(256) void k_mlp(const f16* __restrict__ ynorm,
                                             const f16* __restrict__ b1f,
                                             const f16* __restrict__ b2f,
                                             const float* __restrict__ pw1_b,
                                             const float* __restrict__ pw2_b,
                                             const float* __restrict__ gamma,
                                             float* __restrict__ xt /* BHWC in/out */) {
    __shared__ f16 stage[8][16 * 16];
    const int tid   = threadIdx.x;
    const int lane  = tid & 31;
    const int wv    = tid >> 5;
    const int m0    = blockIdx.x * 128 + wv * 16;
    const int nlane = lane & 15;
    const int khalf = (lane >> 4) * 8;   // A-fragment K sub-offset per ISA layout

    // --- load A1 fragments (16 rows x K=96) : two b128 per 32-wide k-block ---
    v8h a1lo[3], a1hi[3];
    const f16* yrow = ynorm + (size_t)(m0 + nlane) * CC;
#pragma unroll
    for (int kb = 0; kb < 3; ++kb) {
        a1lo[kb] = *(const v8h*)(yrow + kb * 32 + khalf);        // K elems e<8  -> k+e
        a1hi[kb] = *(const v8h*)(yrow + kb * 32 + khalf + 16);   // K elems e>=8 -> k+16+e
    }

    v8h a2lo[12], a2hi[12];
    f16* st = stage[wv];

    // --- GEMM1 + GELU, building GEMM2 A fragments through LDS ---
    for (int nt = 0; nt < 24; ++nt) {
        v8f acc = {};
#pragma unroll
        for (int kb = 0; kb < 3; ++kb) {
            const f16* bp = b1f + ((size_t)(nt * 3 + kb) * 32 + lane) * 16;
            v16h bv = cat16(*(const v8h*)bp, *(const v8h*)(bp + 8));
            acc = wmma16(cat16(a1lo[kb], a1hi[kb]), bv, acc);
        }
        const float bias = pw1_b[nt * 16 + nlane];
        const int mloc = (lane >> 4) * 8;   // C layout: lanes>=16 hold rows 8..15
#pragma unroll
        for (int r = 0; r < 8; ++r) {
            float v = acc[r] + bias;
            st[(mloc + r) * 16 + nlane] = (f16)gelu_f(v);
        }
        __syncthreads();
        v8h rd = *(const v8h*)(st + nlane * 16 + khalf);   // row=nlane, K-offset=khalf..+7
        if (nt & 1) a2hi[nt >> 1] = rd; else a2lo[nt >> 1] = rd;
        __syncthreads();
    }

    // --- GEMM2 + bias + gamma + residual ---
    for (int nt2 = 0; nt2 < 6; ++nt2) {
        v8f acc = {};
#pragma unroll
        for (int kb = 0; kb < 12; ++kb) {
            const f16* bp = b2f + ((size_t)(nt2 * 12 + kb) * 32 + lane) * 16;
            v16h bv = cat16(*(const v8h*)bp, *(const v8h*)(bp + 8));
            acc = wmma16(cat16(a2lo[kb], a2hi[kb]), bv, acc);
        }
        const int   c  = nt2 * 16 + nlane;
        const float bb = pw2_b[c];
        const float gl = gamma[c];
        const int mbase = m0 + (lane >> 4) * 8;
#pragma unroll
        for (int r = 0; r < 8; ++r) {
            size_t idx = (size_t)(mbase + r) * CC + c;
            xt[idx] = xt[idx] + gl * (acc[r] + bb);
        }
    }
}

// ---------------------------------------------------------------------------
extern "C" void kernel_launch(void* const* d_in, const int* in_sizes, int n_in,
                              void* d_out, int out_size, void* d_ws, size_t ws_size,
                              hipStream_t stream) {
    const float* x      = (const float*)d_in[0];
    const float* lut1   = (const float*)d_in[1];
    const float* lut2   = (const float*)d_in[2];
    const float* dw_w   = (const float*)d_in[3];
    const float* dw_b   = (const float*)d_in[4];
    const float* norm_w = (const float*)d_in[5];
    const float* norm_b = (const float*)d_in[6];
    const float* pw1_w  = (const float*)d_in[7];
    const float* pw1_b  = (const float*)d_in[8];
    const float* pw2_w  = (const float*)d_in[9];
    const float* pw2_b  = (const float*)d_in[10];
    const float* gamma  = (const float*)d_in[11];
    float* out = (float*)d_out;

    char* ws = (char*)d_ws;
    float* xt   = (float*)ws;                                   // BHWC, 12.58 MB (in/out)
    f16*   yn   = (f16*)(ws + (size_t)NPOS * CC * 4);           // 6.29 MB
    f16*   b1f  = (f16*)(ws + (size_t)NPOS * CC * 6);           // 72 KB
    f16*   b2f  = b1f + 24 * 3 * 32 * 16;                       // 72 KB

    // 1) NCHW -> BHWC (per-batch transpose of [96][8192])
    k_transpose<<<dim3(HWS / 32, CC / 32, 4), 256, 0, stream>>>(x, xt, CC, HWS);
    // 2) pack weights into WMMA B fragments (f16)
    k_pack<<<18, 256, 0, stream>>>(pw1_w, pw2_w, b1f, b2f);
    // 3) fused bilinear + depthwise + LayerNorm -> f16 activations
    k_bilin_dw_ln<<<NPOS / 4, 128, 0, stream>>>(xt, lut1, lut2, dw_w, dw_b,
                                                norm_w, norm_b, yn);
    // 4) fused MLP + residual (in-place on BHWC buffer)
    k_mlp<<<NPOS / 128, 256, 0, stream>>>(yn, b1f, b2f, pw1_b, pw2_b, gamma, xt);
    // 5) BHWC -> NCHW into d_out (per-batch transpose of [8192][96])
    k_transpose<<<dim3(CC / 32, HWS / 32, 4), 256, 0, stream>>>(xt, out, HWS, CC);
}
